// RoPEALiBiLatentTransformerDecoderLayer_20194936226125
// MI455X (gfx1250) — matrix operand order repaired and checked by reading, not compiled, not run on hardware
//
#include <hip/hip_runtime.h>

#define D_  1024
#define P_  512
#define H_  16
#define FF_ 4096
#define HD_ 64
#define B_  2
#define L_  1024
#define M_  (B_ * L_)

typedef __attribute__((ext_vector_type(16))) __bf16 v16bf;
typedef __attribute__((ext_vector_type(8)))  float  v8f;
typedef unsigned short u16;
typedef __attribute__((ext_vector_type(8))) unsigned short u16x8;

#if defined(__has_builtin)
#if __has_builtin(__builtin_amdgcn_tensor_load_to_lds) && \
    __has_builtin(__builtin_amdgcn_s_wait_tensorcnt)
#define HAS_TDM 1
#endif
#endif
#ifndef HAS_TDM
#define HAS_TDM 0
#endif

__device__ __forceinline__ u16 f2b(float f) {
  union { float f; unsigned int u; } c; c.f = f;
  unsigned int u = c.u;
  unsigned int r = u + 0x7fffu + ((u >> 16) & 1u);
  return (u16)(r >> 16);
}
__device__ __forceinline__ float b2f(u16 h) {
  union { unsigned int u; float f; } c; c.u = ((unsigned int)h) << 16;
  return c.f;
}
__device__ __forceinline__ v8f v8f_zero() {
  v8f z = {0.f, 0.f, 0.f, 0.f, 0.f, 0.f, 0.f, 0.f};
  return z;
}

union FragBF { v16bf v; unsigned int u[8]; u16 s[16]; };

// K index of the dword pair held in VGPR v for a 16x32 bf16 A-fragment,
// lane-half h (contiguous order k = 16h + 2v, matching the B fragments).
__device__ __forceinline__ int a_kidx(int v, int h) {
  return 16 * h + 2 * v;
}

#if HAS_TDM
typedef __attribute__((ext_vector_type(4))) unsigned int u32x4;
typedef __attribute__((ext_vector_type(4))) int i32x4;
typedef __attribute__((ext_vector_type(8))) int i32x8;

// 2D tile DMA: global (rows x cols bf16, row stride stride_elems) -> LDS at
// lds_byte_off, with 16B of LDS padding appended per row (pad_amount code 3
// = 4 dwords). pad_code: 3 => pad every 16 dwords (64B rows), 4 => every 32
// dwords (128B rows). D# bitfields per CDNA5 ISA ch.8.
__device__ __forceinline__ void tdm_load_2d(unsigned lds_byte_off, const void* g,
                                            unsigned rows, unsigned cols,
                                            unsigned stride_elems,
                                            unsigned pad_code) {
  unsigned long long ga = (unsigned long long)g;
  u32x4 g0 = {1u,                                   // count = 1 valid descriptor
              lds_byte_off,                         // lds_addr
              (unsigned)(ga & 0xffffffffull),       // global_addr[31:0]
              (unsigned)((ga >> 32) & 0x01ffffffull) | (2u << 30)}; // [56:32]+type=2
  i32x8 g1;
  g1[0] = (int)((1u << 16)                 // data_size = 1 (2 bytes)
                | (1u << 20)               // pad_enable
                | (pad_code << 22)         // pad_interval
                | (3u << 25));             // pad_amount = 4 dwords (16B)
  g1[1] = (int)((cols & 0xffffu) << 16);                                // dim0 lo
  g1[2] = (int)(((cols >> 16) & 0xffffu) | ((rows & 0xffffu) << 16));   // dim0 hi | dim1 lo
  g1[3] = (int)(((rows >> 16) & 0xffffu) | ((cols & 0xffffu) << 16));   // dim1 hi | tile0
  g1[4] = (int)(rows & 0xffffu);                                        // tile1 (tile2=0)
  g1[5] = (int)stride_elems;                                            // dim0 stride lo
  g1[6] = 0;
  g1[7] = 0;
  i32x4 gz = {0, 0, 0, 0};
#if __clang_major__ >= 23
  i32x8 gz8 = {0, 0, 0, 0, 0, 0, 0, 0};
  __builtin_amdgcn_tensor_load_to_lds(g0, g1, gz, gz, gz8, 0);
#else
  __builtin_amdgcn_tensor_load_to_lds(g0, g1, gz, gz, 0);
#endif
}
__device__ __forceinline__ unsigned lds_off(const void* p) {
  return (unsigned)(unsigned long long)p;
}
#endif

// ---------------------------------------------------------------------------
// f32 -> bf16 elementwise conversion
// ---------------------------------------------------------------------------
__global__ __launch_bounds__(256) void cvt_kernel(const float* __restrict__ in,
                                                  u16* __restrict__ out, size_t n) {
  size_t i = (size_t)blockIdx.x * 256 + threadIdx.x;
  if (i < n) out[i] = f2b(in[i]);
}

// ---------------------------------------------------------------------------
// f32 [K,N] -> bf16 transposed [N,K] (32x32 LDS tiles, coalesced both sides)
// ---------------------------------------------------------------------------
__global__ __launch_bounds__(256) void cvtT_kernel(const float* __restrict__ in,
                                                   u16* __restrict__ out,
                                                   int K, int N) {
  __shared__ float t[32][33];
  const int k0 = blockIdx.y * 32;
  const int n0 = blockIdx.x * 32;
  const int tx = threadIdx.x & 31;
  const int ty = threadIdx.x >> 5;
  for (int i = ty; i < 32; i += 8)
    t[i][tx] = in[(size_t)(k0 + i) * N + n0 + tx];
  __syncthreads();
  for (int i = ty; i < 32; i += 8)
    out[(size_t)(n0 + i) * K + k0 + tx] = f2b(t[tx][i]);
}

// ---------------------------------------------------------------------------
// bf16 V [B*L, D] (head-interleaved) -> bf16 vT [B*H, HD, L]
// ---------------------------------------------------------------------------
__global__ __launch_bounds__(256) void vtrans_kernel(const u16* __restrict__ v,
                                                     u16* __restrict__ vt) {
  __shared__ u16 t[32][33];
  const int bh = blockIdx.z;
  const int l0 = blockIdx.x * 32;
  const int d0 = blockIdx.y * 32;
  const int b = bh >> 4, h = bh & 15;
  const int tx = threadIdx.x & 31;
  const int ty = threadIdx.x >> 5;
  for (int i = ty; i < 32; i += 8)
    t[i][tx] = v[(size_t)(b * L_ + l0 + i) * D_ + h * HD_ + d0 + tx];
  __syncthreads();
  for (int i = ty; i < 32; i += 8)
    vt[((size_t)bh * HD_ + d0 + i) * L_ + l0 + tx] = t[tx][i];
}

// ---------------------------------------------------------------------------
// In-place RoPE on bf16 [M, D] viewed as [M, H, HD]; pos: [L, HD] f32 angles
// ---------------------------------------------------------------------------
__global__ __launch_bounds__(256) void rope_kernel(u16* __restrict__ X,
                                                   const float* __restrict__ pos) {
  size_t i = (size_t)blockIdx.x * 256 + threadIdx.x;
  if (i >= (size_t)M_ * H_ * 32) return;
  int d  = (int)(i & 31);
  int hh = (int)((i >> 5) & (H_ - 1));
  size_t m = i >> 9;
  int l = (int)(m & (L_ - 1));
  float pa = pos[l * HD_ + d];
  float pb = pos[l * HD_ + d + 32];
  size_t base = m * D_ + (size_t)hh * HD_;
  float xa = b2f(X[base + d]);
  float xb = b2f(X[base + d + 32]);
  X[base + d]      = f2b(xa * cosf(pa) - xb * sinf(pa));
  X[base + d + 32] = f2b(xb * cosf(pb) + xa * sinf(pb));
}

// ---------------------------------------------------------------------------
// bf16 GEMM: C[M,N] = A[M,K] @ W[K,N] + bias, optional exact GELU.
// Wt is W transposed: [N,K]. Block tile 64x128, 8 waves, wave = 32x32.
// TDM double-buffered tile loads (fallback: cooperative vector loads).
// ---------------------------------------------------------------------------
__global__ __launch_bounds__(256) void gemm_bf16_kernel(
    const u16* __restrict__ A, const u16* __restrict__ Wt,
    const float* __restrict__ bias,
    float* __restrict__ outF, u16* __restrict__ outB,
    int M, int N, int K, int act) {
  __shared__ __align__(16) u16 lA[2][64][40];    // 32 + 8 pad
  __shared__ __align__(16) u16 lB[2][128][40];   // transposed tile, 32 + 8 pad
  const int bm = blockIdx.y * 64;
  const int bn = blockIdx.x * 128;
  const int tid = threadIdx.x;
  const int wave = tid >> 5;
  const int lane = tid & 31;
  const int lmod = lane & 15;
  const int lhalf = lane >> 4;
  const int wm = (wave >> 2) * 32;
  const int wn = (wave & 3) * 32;
  const int nt = K / 32;

  v8f acc[2][2];
  for (int i = 0; i < 2; ++i)
    for (int j = 0; j < 2; ++j) acc[i][j] = v8f_zero();

#if HAS_TDM
  if (wave == 0) {
    tdm_load_2d(lds_off(&lA[0][0][0]), &A[(size_t)bm * K], 64u, 32u, (unsigned)K, 3u);
    tdm_load_2d(lds_off(&lB[0][0][0]), &Wt[(size_t)bn * K], 128u, 32u, (unsigned)K, 3u);
  }
#else
  const int ar = tid >> 2, ac = (tid & 3) * 8;   // A: 64 rows x 32 k
  const int br = tid >> 1, bc = (tid & 1) * 16;  // Wt: 128 rows x 32 k
#endif

  for (int it = 0; it < nt; ++it) {
    const int p = it & 1;
#if HAS_TDM
    if (wave == 0) {
      if (it + 1 < nt) {
        tdm_load_2d(lds_off(&lA[1 - p][0][0]),
                    &A[(size_t)bm * K + (it + 1) * 32], 64u, 32u, (unsigned)K, 3u);
        tdm_load_2d(lds_off(&lB[1 - p][0][0]),
                    &Wt[(size_t)bn * K + (it + 1) * 32], 128u, 32u, (unsigned)K, 3u);
        __builtin_amdgcn_s_wait_tensorcnt(2);
      } else {
        __builtin_amdgcn_s_wait_tensorcnt(0);
      }
    }
#else
    {
      int k0 = it * 32;
      *(u16x8*)&lA[p][ar][ac]     = *(const u16x8*)&A [(size_t)(bm + ar) * K + k0 + ac];
      *(u16x8*)&lB[p][br][bc]     = *(const u16x8*)&Wt[(size_t)(bn + br) * K + k0 + bc];
      *(u16x8*)&lB[p][br][bc + 8] = *(const u16x8*)&Wt[(size_t)(bn + br) * K + k0 + bc + 8];
    }
#endif
    __syncthreads();   // tile it visible to all waves

    FragBF af[2];
#pragma unroll
    for (int i = 0; i < 2; ++i) {
      int m = wm + i * 16 + lmod;
#pragma unroll
      for (int v = 0; v < 8; ++v)
        af[i].u[v] = *(const unsigned int*)&lA[p][m][a_kidx(v, lhalf)];
    }
    FragBF bfm[2];
#pragma unroll
    for (int j = 0; j < 2; ++j) {
      int n = wn + j * 16 + lmod;
#pragma unroll
      for (int v = 0; v < 8; ++v)
        bfm[j].u[v] = *(const unsigned int*)&lB[p][n][16 * lhalf + 2 * v];
    }
#pragma unroll
    for (int i = 0; i < 2; ++i)
#pragma unroll
      for (int j = 0; j < 2; ++j)
        acc[i][j] = __builtin_amdgcn_wmma_f32_16x16x32_bf16(
            false, af[i].v, false, bfm[j].v, (short)0, acc[i][j], false, false);
    __syncthreads();   // reads done; next iteration may overwrite other buffer
  }

#pragma unroll
  for (int i = 0; i < 2; ++i) {
#pragma unroll
    for (int j = 0; j < 2; ++j) {
      int n = bn + wn + j * 16 + lmod;
      float bval = bias[n];
#pragma unroll
      for (int r = 0; r < 8; ++r) {
        int m = bm + wm + i * 16 + r + 8 * lhalf;
        float x = acc[i][j][r] + bval;
        if (act == 1) x = 0.5f * x * (1.0f + erff(x * 0.70710678118654752f));
        size_t idx = (size_t)m * N + n;
        if (outF) outF[idx] = x;
        if (outB) outB[idx] = f2b(x);
      }
    }
  }
}

// ---------------------------------------------------------------------------
// Fused flash attention (bf16 QKV, f32 softmax, WMMA for QK^T and PV).
// Grid: (L/128, B*H). Each wave owns 16 query rows. V consumed transposed.
// scores = (QK^T)*0.125 + alibi[h,q,k], masked keys -> -1e30.
// TDM double-buffered K/V tile loads (fallback: cooperative vector loads).
// ---------------------------------------------------------------------------
__global__ __launch_bounds__(256) void attn_kernel(
    const u16* __restrict__ Q, const u16* __restrict__ Kk,
    const u16* __restrict__ Vt, const float* __restrict__ alibi,
    const int* __restrict__ mask, u16* __restrict__ Out) {
  __shared__ __align__(16) u16 lQ[128][72];        // 64 + 8 pad
  __shared__ __align__(16) u16 lK[2][64][72];
  __shared__ __align__(16) u16 lVt[2][64][72];     // [d][key]
  __shared__ __align__(16) u16 lP[8][16][72];

  const int bh = blockIdx.y;
  const int b = bh / H_;
  const int h = bh % H_;
  const int q0 = blockIdx.x * 128;
  const int tid = threadIdx.x;
  const int wave = tid >> 5;
  const int lane = tid & 31;
  const int lmod = lane & 15;
  const int lhalf = lane >> 4;
  const int wq = wave * 16;
  const int NT = L_ / 64;

  {
    int r = tid >> 1;
    int c = (tid & 1) * 32;
    const u16* src = &Q[((size_t)(b * L_ + q0 + r)) * D_ + h * HD_ + c];
#pragma unroll
    for (int j = 0; j < 4; ++j)
      *(u16x8*)&lQ[r][c + j * 8] = *(const u16x8*)(src + j * 8);
  }
#if HAS_TDM
  if (wave == 0) {
    tdm_load_2d(lds_off(&lK[0][0][0]),  &Kk[((size_t)b * L_) * D_ + h * HD_],
                64u, 64u, (unsigned)D_, 4u);
    tdm_load_2d(lds_off(&lVt[0][0][0]), &Vt[(size_t)bh * HD_ * L_],
                64u, 64u, (unsigned)L_, 4u);
  }
#else
  const int kr = tid >> 2, kc = (tid & 3) * 16;
#endif
  __syncthreads();

  FragBF aQ[2];
#pragma unroll
  for (int ks = 0; ks < 2; ++ks) {
    int m = wq + lmod;
#pragma unroll
    for (int v = 0; v < 8; ++v)
      aQ[ks].u[v] = *(const unsigned int*)&lQ[m][a_kidx(v, lhalf) + ks * 32];
  }

  v8f oacc[4];
#pragma unroll
  for (int db = 0; db < 4; ++db) oacc[db] = v8f_zero();
  float mO[8], lS[8];
#pragma unroll
  for (int r = 0; r < 8; ++r) { mO[r] = -3.0e38f; lS[r] = 0.f; }

  for (int kt = 0; kt < NT; ++kt) {
    const int p = kt & 1;
#if HAS_TDM
    if (wave == 0) {
      if (kt + 1 < NT) {
        tdm_load_2d(lds_off(&lK[1 - p][0][0]),
                    &Kk[((size_t)(b * L_ + (kt + 1) * 64)) * D_ + h * HD_],
                    64u, 64u, (unsigned)D_, 4u);
        tdm_load_2d(lds_off(&lVt[1 - p][0][0]),
                    &Vt[(size_t)bh * HD_ * L_ + (kt + 1) * 64],
                    64u, 64u, (unsigned)L_, 4u);
        __builtin_amdgcn_s_wait_tensorcnt(2);
      } else {
        __builtin_amdgcn_s_wait_tensorcnt(0);
      }
    }
#else
    {
      const u16* sk = &Kk[((size_t)(b * L_ + kt * 64 + kr)) * D_ + h * HD_ + kc];
      const u16* sv = &Vt[((size_t)bh * HD_ + kr) * L_ + kt * 64 + kc];
      *(u16x8*)&lK[p][kr][kc]      = *(const u16x8*)sk;
      *(u16x8*)&lK[p][kr][kc + 8]  = *(const u16x8*)(sk + 8);
      *(u16x8*)&lVt[p][kr][kc]     = *(const u16x8*)sv;
      *(u16x8*)&lVt[p][kr][kc + 8] = *(const u16x8*)(sv + 8);
    }
#endif
    __syncthreads();   // tile kt visible

    v8f s[4];
#pragma unroll
    for (int kb = 0; kb < 4; ++kb) {
      s[kb] = v8f_zero();
#pragma unroll
      for (int ks = 0; ks < 2; ++ks) {
        FragBF bK;
        int n = kb * 16 + lmod;
#pragma unroll
        for (int v = 0; v < 8; ++v)
          bK.u[v] = *(const unsigned int*)&lK[p][n][ks * 32 + 16 * lhalf + 2 * v];
        s[kb] = __builtin_amdgcn_wmma_f32_16x16x32_bf16(
            false, aQ[ks].v, false, bK.v, (short)0, s[kb], false, false);
      }
    }

#pragma unroll
    for (int kb = 0; kb < 4; ++kb) {
      int kcol = kt * 64 + kb * 16 + lmod;
      int mk = mask[b * L_ + kcol];
#pragma unroll
      for (int r = 0; r < 8; ++r) {
        int qrow = q0 + wq + r + 8 * lhalf;
        float val = s[kb][r] * 0.125f + alibi[((size_t)h * L_ + qrow) * L_ + kcol];
        if (mk) val = -1.0e30f;
        s[kb][r] = val;
      }
    }

    float mnew[8], scl[8], psum[8];
#pragma unroll
    for (int r = 0; r < 8; ++r) {
      float rm = fmaxf(fmaxf(s[0][r], s[1][r]), fmaxf(s[2][r], s[3][r]));
#pragma unroll
      for (int off = 8; off >= 1; off >>= 1)
        rm = fmaxf(rm, __shfl_xor(rm, off, 32));
      mnew[r] = fmaxf(mO[r], rm);
      scl[r] = __expf(mO[r] - mnew[r]);
      psum[r] = 0.f;
    }
#pragma unroll
    for (int kb = 0; kb < 4; ++kb) {
#pragma unroll
      for (int r = 0; r < 8; ++r) {
        float pv = __expf(s[kb][r] - mnew[r]);
        psum[r] += pv;
        lP[wave][r + 8 * lhalf][kb * 16 + lmod] = f2b(pv);
      }
    }
#pragma unroll
    for (int r = 0; r < 8; ++r) {
      float ps = psum[r];
#pragma unroll
      for (int off = 8; off >= 1; off >>= 1) ps += __shfl_xor(ps, off, 32);
      lS[r] = lS[r] * scl[r] + ps;
      mO[r] = mnew[r];
    }
#pragma unroll
    for (int db = 0; db < 4; ++db)
#pragma unroll
      for (int r = 0; r < 8; ++r) oacc[db][r] *= scl[r];

    __syncthreads();   // lP visible (and lK reads done)

#pragma unroll
    for (int ks = 0; ks < 2; ++ks) {
      FragBF aP;
#pragma unroll
      for (int v = 0; v < 8; ++v)
        aP.u[v] = *(const unsigned int*)&lP[wave][lmod][a_kidx(v, lhalf) + ks * 32];
#pragma unroll
      for (int db = 0; db < 4; ++db) {
        FragBF bV;
        int n = db * 16 + lmod;
#pragma unroll
        for (int v = 0; v < 8; ++v)
          bV.u[v] = *(const unsigned int*)&lVt[p][n][ks * 32 + 16 * lhalf + 2 * v];
        oacc[db] = __builtin_amdgcn_wmma_f32_16x16x32_bf16(
            false, aP.v, false, bV.v, (short)0, oacc[db], false, false);
      }
    }
    __syncthreads();   // all reads done before next tile overwrites buffers
  }

#pragma unroll
  for (int db = 0; db < 4; ++db) {
    int n = h * HD_ + db * 16 + lmod;
#pragma unroll
    for (int r = 0; r < 8; ++r) {
      int m = b * L_ + q0 + wq + r + 8 * lhalf;
      Out[(size_t)m * D_ + n] = f2b(oacc[db][r] / lS[r]);
    }
  }
}

// ---------------------------------------------------------------------------
// Residual + LayerNorm over D=1024. mode 0: out = res + LN(y); mode 1: out = LN(res + y)
// ---------------------------------------------------------------------------
__global__ __launch_bounds__(256) void ln_kernel(
    const float* __restrict__ res, const float* __restrict__ y,
    const float* __restrict__ g, const float* __restrict__ be,
    float* __restrict__ outF, u16* __restrict__ outB, int mode) {
  const int row = blockIdx.x;
  const int t = threadIdx.x;
  __shared__ float sred[8];
  float v[4];
  float s = 0.f;
#pragma unroll
  for (int i = 0; i < 4; ++i) {
    int c = t + i * 256;
    float val = y[(size_t)row * D_ + c];
    if (mode == 1) val += res[(size_t)row * D_ + c];
    v[i] = val;
    s += val;
  }
#pragma unroll
  for (int off = 16; off >= 1; off >>= 1) s += __shfl_xor(s, off, 32);
  if ((t & 31) == 0) sred[t >> 5] = s;
  __syncthreads();
  if (t == 0) {
    float tot = 0.f;
    for (int i = 0; i < 8; ++i) tot += sred[i];
    sred[0] = tot;
  }
  __syncthreads();
  float mu = sred[0] * (1.0f / D_);
  __syncthreads();
  float s2 = 0.f;
#pragma unroll
  for (int i = 0; i < 4; ++i) { float d = v[i] - mu; s2 += d * d; }
#pragma unroll
  for (int off = 16; off >= 1; off >>= 1) s2 += __shfl_xor(s2, off, 32);
  if ((t & 31) == 0) sred[t >> 5] = s2;
  __syncthreads();
  if (t == 0) {
    float tot = 0.f;
    for (int i = 0; i < 8; ++i) tot += sred[i];
    sred[0] = tot;
  }
  __syncthreads();
  float rstd = rsqrtf(sred[0] * (1.0f / D_) + 1e-5f);
#pragma unroll
  for (int i = 0; i < 4; ++i) {
    int c = t + i * 256;
    float ln = (v[i] - mu) * rstd * g[c] + be[c];
    float o = (mode == 0) ? res[(size_t)row * D_ + c] + ln : ln;
    size_t idx = (size_t)row * D_ + c;
    outF[idx] = o;
    if (outB) outB[idx] = f2b(o);
  }
}

// ---------------------------------------------------------------------------
extern "C" void kernel_launch(void* const* d_in, const int* in_sizes, int n_in,
                              void* d_out, int out_size, void* d_ws, size_t ws_size,
                              hipStream_t stream) {
  (void)in_sizes; (void)n_in; (void)out_size; (void)ws_size;
  const float* memory_f    = (const float*)d_in[0];
  const float* tgt_f       = (const float*)d_in[1];
  const float* self_alibi  = (const float*)d_in[2];
  const float* self_pos    = (const float*)d_in[3];
  const float* cross_alibi = (const float*)d_in[4];
  const float* cross_pos   = (const float*)d_in[5];
  const int*   mask        = (const int*)d_in[6];
  const float* sa_lat_w = (const float*)d_in[7];  const float* sa_lat_b = (const float*)d_in[8];
  const float* sa_q_w   = (const float*)d_in[9];  const float* sa_q_b   = (const float*)d_in[10];
  const float* sa_k_w   = (const float*)d_in[11]; const float* sa_k_b   = (const float*)d_in[12];
  const float* sa_v_w   = (const float*)d_in[13]; const float* sa_v_b   = (const float*)d_in[14];
  const float* sa_out_w = (const float*)d_in[15]; const float* sa_out_b = (const float*)d_in[16];
  const float* ca_lat_w = (const float*)d_in[17]; const float* ca_lat_b = (const float*)d_in[18];
  const float* ca_q_w   = (const float*)d_in[19]; const float* ca_q_b   = (const float*)d_in[20];
  const float* ca_k_w   = (const float*)d_in[21]; const float* ca_k_b   = (const float*)d_in[22];
  const float* ca_v_w   = (const float*)d_in[23]; const float* ca_v_b   = (const float*)d_in[24];
  const float* ca_out_w = (const float*)d_in[25]; const float* ca_out_b = (const float*)d_in[26];
  const float* lin1_w   = (const float*)d_in[27]; const float* lin1_b   = (const float*)d_in[28];
  const float* lin2_w   = (const float*)d_in[29]; const float* lin2_b   = (const float*)d_in[30];
  const float* n1_g = (const float*)d_in[31]; const float* n1_b = (const float*)d_in[32];
  const float* n2_g = (const float*)d_in[33]; const float* n2_b = (const float*)d_in[34];
  const float* n3_g = (const float*)d_in[35]; const float* n3_b = (const float*)d_in[36];
  float* out = (float*)d_out;

  char* base = (char*)d_ws;
  size_t off = 0;
  auto allocB = [&](size_t elems) -> u16* {
    u16* p = (u16*)(base + off);
    off = (off + elems * sizeof(u16) + 255) & ~(size_t)255;
    return p;
  };
  auto allocF = [&](size_t elems) -> float* {
    float* p = (float*)(base + off);
    off = (off + elems * sizeof(float) + 255) & ~(size_t)255;
    return p;
  };

  // weights stored transposed [N,K] in bf16
  u16* wb_sa_lat = allocB((size_t)D_ * P_);
  u16* wb_sa_q   = allocB((size_t)P_ * D_);
  u16* wb_sa_k   = allocB((size_t)P_ * D_);
  u16* wb_sa_v   = allocB((size_t)P_ * D_);
  u16* wb_sa_out = allocB((size_t)D_ * D_);
  u16* wb_ca_lat = allocB((size_t)D_ * P_);
  u16* wb_ca_q   = allocB((size_t)D_ * D_);
  u16* wb_ca_k   = allocB((size_t)P_ * D_);
  u16* wb_ca_v   = allocB((size_t)P_ * D_);
  u16* wb_ca_out = allocB((size_t)D_ * D_);
  u16* wb_lin1   = allocB((size_t)D_ * FF_);
  u16* wb_lin2   = allocB((size_t)FF_ * D_);
  u16* tgt_bf  = allocB((size_t)M_ * D_);
  u16* mem_bf  = allocB((size_t)M_ * D_);
  u16* proj_bf = allocB((size_t)M_ * P_);
  u16* q_bf    = allocB((size_t)M_ * D_);
  u16* k_bf    = allocB((size_t)M_ * D_);
  u16* v_bf    = allocB((size_t)M_ * D_);
  u16* vt_bf   = allocB((size_t)M_ * D_);
  u16* attn_bf = allocB((size_t)M_ * D_);
  u16* x_bf    = allocB((size_t)M_ * D_);
  u16* f1_bf   = allocB((size_t)M_ * FF_);
  float* y_f = allocF((size_t)M_ * D_);
  float* x_f = allocF((size_t)M_ * D_);

  auto cvt = [&](const float* src, u16* dst, size_t n) {
    cvt_kernel<<<(unsigned)((n + 255) / 256), 256, 0, stream>>>(src, dst, n);
  };
  auto cvtT = [&](const float* src, u16* dst, int K, int N) {
    dim3 grid(N / 32, K / 32);
    cvtT_kernel<<<grid, 256, 0, stream>>>(src, dst, K, N);
  };
  auto gemm = [&](const u16* A, const u16* Wt, const float* bias,
                  float* oF, u16* oB, int M, int N, int K, int act) {
    dim3 grid(N / 128, M / 64);
    gemm_bf16_kernel<<<grid, 256, 0, stream>>>(A, Wt, bias, oF, oB, M, N, K, act);
  };
  auto vtrans = [&]() {
    dim3 grid(L_ / 32, HD_ / 32, B_ * H_);
    vtrans_kernel<<<grid, 256, 0, stream>>>(v_bf, vt_bf);
  };
  auto rope = [&](u16* X, const float* pos) {
    rope_kernel<<<(M_ * H_ * 32) / 256, 256, 0, stream>>>(X, pos);
  };
  auto attn = [&](const float* alibi) {
    dim3 grid(L_ / 128, B_ * H_);
    attn_kernel<<<grid, 256, 0, stream>>>(q_bf, k_bf, vt_bf, alibi, mask, attn_bf);
  };
  auto ln = [&](const float* res, const float* y, const float* g, const float* be,
                float* oF, u16* oB, int mode) {
    ln_kernel<<<M_, 256, 0, stream>>>(res, y, g, be, oF, oB, mode);
  };

  // weight + activation conversions (weights transposed to [N,K])
  cvtT(sa_lat_w, wb_sa_lat, D_, P_);
  cvtT(sa_q_w,   wb_sa_q,   P_, D_);
  cvtT(sa_k_w,   wb_sa_k,   P_, D_);
  cvtT(sa_v_w,   wb_sa_v,   P_, D_);
  cvtT(sa_out_w, wb_sa_out, D_, D_);
  cvtT(ca_lat_w, wb_ca_lat, D_, P_);
  cvtT(ca_q_w,   wb_ca_q,   D_, D_);
  cvtT(ca_k_w,   wb_ca_k,   P_, D_);
  cvtT(ca_v_w,   wb_ca_v,   P_, D_);
  cvtT(ca_out_w, wb_ca_out, D_, D_);
  cvtT(lin1_w,   wb_lin1,   D_, FF_);
  cvtT(lin2_w,   wb_lin2,   FF_, D_);
  cvt(tgt_f,    tgt_bf, (size_t)M_ * D_);
  cvt(memory_f, mem_bf, (size_t)M_ * D_);

  // ---- self attention ----
  gemm(tgt_bf, wb_sa_lat, sa_lat_b, nullptr, proj_bf, M_, P_, D_, 0);
  gemm(proj_bf, wb_sa_q, sa_q_b, nullptr, q_bf, M_, D_, P_, 0);
  gemm(proj_bf, wb_sa_k, sa_k_b, nullptr, k_bf, M_, D_, P_, 0);
  gemm(proj_bf, wb_sa_v, sa_v_b, nullptr, v_bf, M_, D_, P_, 0);
  rope(q_bf, self_pos);
  rope(k_bf, self_pos);
  vtrans();
  attn(self_alibi);
  gemm(attn_bf, wb_sa_out, sa_out_b, y_f, nullptr, M_, D_, D_, 0);
  ln(tgt_f, y_f, n1_g, n1_b, x_f, x_bf, 0);

  // ---- cross attention ----
  gemm(mem_bf, wb_ca_lat, ca_lat_b, nullptr, proj_bf, M_, P_, D_, 0);
  gemm(x_bf, wb_ca_q, ca_q_b, nullptr, q_bf, M_, D_, D_, 0);
  gemm(proj_bf, wb_ca_k, ca_k_b, nullptr, k_bf, M_, D_, P_, 0);
  gemm(proj_bf, wb_ca_v, ca_v_b, nullptr, v_bf, M_, D_, P_, 0);
  rope(q_bf, cross_pos);
  rope(k_bf, cross_pos);
  vtrans();
  attn(cross_alibi);
  gemm(attn_bf, wb_ca_out, ca_out_b, y_f, nullptr, M_, D_, D_, 0);
  ln(x_f, y_f, n2_g, n2_b, x_f, x_bf, 0);

  // ---- FFN ----
  gemm(x_bf, wb_lin1, lin1_b, nullptr, f1_bf, M_, FF_, D_, 1);
  gemm(f1_bf, wb_lin2, lin2_b, y_f, nullptr, M_, D_, FF_, 0);
  ln(x_f, y_f, n3_g, n3_b, out, nullptr, 1);
}
